// NormEMAVectorQuantizer_91336774516844
// MI455X (gfx1250) — compile-verified, matrix operand order checked by reading, not run on hardware
//
#include <hip/hip_runtime.h>
#include <hip/hip_bf16.h>
#include <math.h>

#define KCODES 8192
#define PLEN   128
#define HDIM   16
#define BATCH  16
#define TLEN   8192
#define VCH    7
#define LPP    64                    // TLEN / PLEN
#define NROWS  (BATCH * VCH * LPP)   // 7168
#define NGROUP (NROWS / 16)          // 448 wave-groups
#define DECAYF 0.99f

typedef __attribute__((ext_vector_type(16))) _Float16 v16h;
typedef __attribute__((ext_vector_type(8)))  float    v8f;

// ---------------------------------------------------------------------------
// 1) |FFT128| of each non-overlapping patch.  x is [B, T, V]; patch (b,v,lp)
//    reads x[b, lp*128+t, v].  Twiddles cached in LDS; exp periodic mod 128.
// ---------------------------------------------------------------------------
__global__ void __launch_bounds__(128) fft_abs_kernel(const float* __restrict__ x,
                                                      float* __restrict__ z) {
  __shared__ float s[PLEN], twc[PLEN], tws[PLEN];
  const int p   = blockIdx.x;            // (b*7+v)*64 + lp
  const int t   = threadIdx.x;
  const int b   = p / (VCH * LPP);
  const int v   = (p / LPP) % VCH;
  const int lp  = p % LPP;
  s[t] = x[((size_t)b * TLEN + (size_t)lp * PLEN + t) * VCH + v];
  float sv, cv;
  __sincosf(-(float)t * (6.283185307179586f / 128.0f), &sv, &cv);
  twc[t] = cv; tws[t] = sv;
  __syncthreads();
  float re = 0.f, im = 0.f;
#pragma unroll 4
  for (int n = 0; n < PLEN; ++n) {
    const int j = (t * n) & 127;
    re = fmaf(s[n], twc[j], re);
    im = fmaf(s[n], tws[j], im);
  }
  z[(size_t)p * PLEN + t] = sqrtf(re * re + im * im);
}

// ---------------------------------------------------------------------------
// 2) Per-Lp mean and 1/(std+1e-5)   (ddof=1 over B*V*P = 14336 samples)
// ---------------------------------------------------------------------------
__global__ void __launch_bounds__(256) stats_kernel(const float* __restrict__ z,
                                                    float* __restrict__ meanv,
                                                    float* __restrict__ rstdv) {
  __shared__ float ss[256], sq[256];
  const int lp = blockIdx.x, tid = threadIdx.x;
  float a = 0.f, b = 0.f;
  for (int j = tid; j < BATCH * VCH * PLEN; j += 256) {
    const int row = j >> 7, t = j & 127;
    const float val = z[((size_t)row * LPP + lp) * PLEN + t];
    a += val; b = fmaf(val, val, b);
  }
  ss[tid] = a; sq[tid] = b;
  __syncthreads();
  for (int st = 128; st > 0; st >>= 1) {
    if (tid < st) { ss[tid] += ss[tid + st]; sq[tid] += sq[tid + st]; }
    __syncthreads();
  }
  if (tid == 0) {
    const float N = (float)(BATCH * VCH * PLEN);
    const float m = ss[0] / N;
    float var = (sq[0] - N * m * m) / (N - 1.0f);
    var = fmaxf(var, 0.0f);
    meanv[lp] = m;
    rstdv[lp] = 1.0f / (sqrtf(var) + 1e-5f);
  }
}

// ---------------------------------------------------------------------------
// 3) Encoder MLP per row, l2norm, emit f16 z_h (only consumer is the WMMA cdist)
// ---------------------------------------------------------------------------
__global__ void __launch_bounds__(128) encoder_kernel(
    const float* __restrict__ z, const float* __restrict__ meanv, const float* __restrict__ rstdv,
    const float* __restrict__ w1, const float* __restrict__ b1,
    const float* __restrict__ w2, const float* __restrict__ b2,
    const float* __restrict__ w3, const float* __restrict__ b3,
    const float* __restrict__ r0wa, const float* __restrict__ r0ba,
    const float* __restrict__ r0wb, const float* __restrict__ r0bb,
    const float* __restrict__ r1wa, const float* __restrict__ r1ba,
    const float* __restrict__ r1wb, const float* __restrict__ r1bb,
    const float* __restrict__ wsc, const float* __restrict__ bsc,
    _Float16* __restrict__ zh16) {
  __shared__ float zn[128], h1[64], h2[32], h3[16], tb[64];
  __shared__ float rn;
  const int r = blockIdx.x, tid = threadIdx.x;
  const int lp = r & 63;
  zn[tid] = (z[(size_t)r * PLEN + tid] - meanv[lp]) * rstdv[lp];
  __syncthreads();
  if (tid < 64) { float s = b1[tid]; for (int n = 0; n < 128; ++n) s = fmaf(zn[n], w1[n * 64 + tid], s); h1[tid] = fmaxf(s, 0.f); }
  __syncthreads();
  if (tid < 32) { float s = b2[tid]; for (int n = 0; n < 64; ++n) s = fmaf(h1[n], w2[n * 32 + tid], s); h2[tid] = fmaxf(s, 0.f); }
  __syncthreads();
  if (tid < 16) { float s = b3[tid]; for (int n = 0; n < 32; ++n) s = fmaf(h2[n], w3[n * 16 + tid], s); h3[tid] = s; }
  __syncthreads();
  // resblock 0:  h += Lb(relu(La(relu(h))))
  if (tid < 64) { float s = r0ba[tid]; for (int n = 0; n < 16; ++n) s = fmaf(fmaxf(h3[n], 0.f), r0wa[n * 64 + tid], s); tb[tid] = fmaxf(s, 0.f); }
  __syncthreads();
  if (tid < 16) { float s = r0bb[tid]; for (int o = 0; o < 64; ++o) s = fmaf(tb[o], r0wb[o * 16 + tid], s); h3[tid] += s; }
  __syncthreads();
  // resblock 1
  if (tid < 64) { float s = r1ba[tid]; for (int n = 0; n < 16; ++n) s = fmaf(fmaxf(h3[n], 0.f), r1wa[n * 64 + tid], s); tb[tid] = fmaxf(s, 0.f); }
  __syncthreads();
  if (tid < 16) { float s = r1bb[tid]; for (int o = 0; o < 64; ++o) s = fmaf(tb[o], r1wb[o * 16 + tid], s); h3[tid] += s; }
  __syncthreads();
  // relu + input shortcut
  if (tid < 16) {
    float s = bsc[tid];
    for (int n = 0; n < 128; ++n) s = fmaf(zn[n], wsc[n * 16 + tid], s);
    h3[tid] = fmaxf(h3[tid], 0.f) + s;
  }
  __syncthreads();
  if (tid == 0) { float s = 0.f; for (int n = 0; n < 16; ++n) s = fmaf(h3[n], h3[n], s); rn = 1.0f / fmaxf(sqrtf(s), 1e-12f); }
  __syncthreads();
  if (tid < 16) zh16[(size_t)r * HDIM + tid] = (_Float16)(h3[tid] * rn);
}

// ---------------------------------------------------------------------------
// 4) Codebook prep: f16 copy + 0.5*|cb|^2
// ---------------------------------------------------------------------------
__global__ void __launch_bounds__(256) cbprep_kernel(const float* __restrict__ cb,
                                                     _Float16* __restrict__ cb16,
                                                     float* __restrict__ cbn2h) {
  const int j = blockIdx.x * 256 + threadIdx.x;
  if (j >= KCODES) return;
  float s = 0.f;
#pragma unroll
  for (int k = 0; k < 16; ++k) {
    const float v = cb[(size_t)j * 16 + k];
    s = fmaf(v, v, s);
    cb16[(size_t)j * 16 + k] = (_Float16)v;
  }
  cbn2h[j] = 0.5f * s;
}

// ---------------------------------------------------------------------------
// 5) cdist + argmin via V_WMMA_F32_16X16X32_F16.
//    Since |z_h|=1:  argmin d = argmax( z_h . cb  -  0.5*|cb|^2 ).
//    K=16 zero-padded to 32 -> ONE wmma per 16x16 score tile.
//    Wave owns 16 rows; 448 groups = 56 blocks x 8 waves (exact -> EXEC all-1s).
//    Two N-tiles per iteration: the second WMMA + its B-fragment load fill the
//    first accumulator's WMMA->VALU hazard window (4 coexec nops otherwise).
//    Within a lane ncol is strictly increasing, so a strict '>' keeps the
//    first (lowest-index) maximum among exact ties; the index tie-break is
//    only needed in the cross-lane reduction.
//    A/B fragment layout per ISA 7.12.2 (16-bit 16x32): lane l holds M/N = l&15,
//    VGPR v (v<4): K = 2v + 8*(l>>4) + {0,1}; VGPR 4..7: K>=16 -> zero.
//    D layout: VGPR r <-> M = r + 8*(l>>4), N = n0 + (l&15).
// ---------------------------------------------------------------------------
__global__ void __launch_bounds__(256) cdist_argmin_kernel(
    const _Float16* __restrict__ zh, const _Float16* __restrict__ cb16,
    const float* __restrict__ cbn2h, int* __restrict__ idx_ws,
    float* __restrict__ idx_out) {
  const int lane = threadIdx.x & 31;
  const int wave = threadIdx.x >> 5;
  const int g    = blockIdx.x * 8 + wave;        // 0..447, exact cover
  const int mrow = (g << 4) + (lane & 15);
  const int kb   = (lane >> 4) << 3;             // 0 or 8

  v16h a = {};
  const _Float16* zrow = zh + (size_t)mrow * HDIM + kb;
#pragma unroll
  for (int v = 0; v < 4; ++v) { a[2 * v] = zrow[2 * v]; a[2 * v + 1] = zrow[2 * v + 1]; }

  float best[8];
  int   bidx[8];
#pragma unroll
  for (int r = 0; r < 8; ++r) { best[r] = -3.4e38f; bidx[r] = 0x7fffffff; }

  const int lcol = lane & 15;
  for (int n0 = 0; n0 < KCODES; n0 += 32) {
    if (n0 + 32 < KCODES) __builtin_prefetch(cb16 + (size_t)(n0 + 32) * HDIM, 0, 1);
    const int ncol0 = n0 + lcol;
    const int ncol1 = n0 + 16 + lcol;
    const _Float16* crow0 = cb16 + (size_t)ncol0 * HDIM + kb;
    const _Float16* crow1 = cb16 + (size_t)ncol1 * HDIM + kb;
    v16h bf0 = {}, bf1 = {};
#pragma unroll
    for (int v = 0; v < 4; ++v) {
      bf0[2 * v] = crow0[2 * v]; bf0[2 * v + 1] = crow0[2 * v + 1];
      bf1[2 * v] = crow1[2 * v]; bf1[2 * v + 1] = crow1[2 * v + 1];
    }
    v8f acc0 = {}, acc1 = {};
    acc0 = __builtin_amdgcn_wmma_f32_16x16x32_f16(false, a, false, bf0,
                                                  (short)0, acc0, false, false);
    acc1 = __builtin_amdgcn_wmma_f32_16x16x32_f16(false, a, false, bf1,
                                                  (short)0, acc1, false, false);
    const float hn0 = cbn2h[ncol0];
    const float hn1 = cbn2h[ncol1];
#pragma unroll
    for (int r = 0; r < 8; ++r) {
      const float s0 = acc0[r] - hn0;
      if (s0 > best[r]) { best[r] = s0; bidx[r] = ncol0; }
    }
#pragma unroll
    for (int r = 0; r < 8; ++r) {
      const float s1 = acc1[r] - hn1;
      if (s1 > best[r]) { best[r] = s1; bidx[r] = ncol1; }
    }
  }
  // reduce argmax across the 16 lanes of each half-wave (lowest index on ties)
#pragma unroll
  for (int r = 0; r < 8; ++r) {
    float s = best[r];
    int   i = bidx[r];
#pragma unroll
    for (int m = 1; m < 16; m <<= 1) {
      const float s2 = __shfl_xor(s, m, 32);
      const int   i2 = __shfl_xor(i, m, 32);
      if (s2 > s || (s2 == s && i2 < i)) { s = s2; i = i2; }
    }
    if ((lane & 15) == 0) {
      const int row = (g << 4) + ((lane >> 4) << 3) + r;  // M = r + 8*(lane>=16)
      idx_ws[row]  = i;
      idx_out[row] = (float)i;
    }
  }
}

// ---------------------------------------------------------------------------
// 6) bincount + EMA
// ---------------------------------------------------------------------------
__global__ void __launch_bounds__(256) bincount_kernel(const int* __restrict__ idx_ws,
                                                       float* __restrict__ counts) {
  const int i = blockIdx.x * 256 + threadIdx.x;
  if (i < NROWS) atomicAdd(&counts[idx_ws[i]], 1.0f);
}

__global__ void __launch_bounds__(256) ema_kernel(const float* __restrict__ cs,
                                                  const float* __restrict__ counts,
                                                  float* __restrict__ out) {
  const int j = blockIdx.x * 256 + threadIdx.x;
  if (j < KCODES) out[j] = cs[j] * DECAYF + counts[j] * (1.0f - DECAYF);
}

// ---------------------------------------------------------------------------
// 7) Decoder (input is exactly codebook[idx] due to straight-through)
// ---------------------------------------------------------------------------
__global__ void __launch_bounds__(128) decoder_kernel(
    const int* __restrict__ idx_ws, const float* __restrict__ cb,
    const float* __restrict__ r0wa, const float* __restrict__ r0ba,
    const float* __restrict__ r0wb, const float* __restrict__ r0bb,
    const float* __restrict__ r1wa, const float* __restrict__ r1ba,
    const float* __restrict__ r1wb, const float* __restrict__ r1bb,
    const float* __restrict__ w1, const float* __restrict__ b1,
    const float* __restrict__ w2, const float* __restrict__ b2,
    const float* __restrict__ w3, const float* __restrict__ b3,
    const float* __restrict__ wsc, const float* __restrict__ bsc,
    float* __restrict__ out) {
  __shared__ float zq[16], h[16], tb[64], h1[32], h2[64];
  const int r = blockIdx.x, tid = threadIdx.x;
  if (tid < 16) { const float v = cb[(size_t)idx_ws[r] * HDIM + tid]; zq[tid] = v; h[tid] = v; }
  __syncthreads();
  if (tid < 64) { float s = r0ba[tid]; for (int n = 0; n < 16; ++n) s = fmaf(fmaxf(h[n], 0.f), r0wa[n * 64 + tid], s); tb[tid] = fmaxf(s, 0.f); }
  __syncthreads();
  if (tid < 16) { float s = r0bb[tid]; for (int o = 0; o < 64; ++o) s = fmaf(tb[o], r0wb[o * 16 + tid], s); h[tid] += s; }
  __syncthreads();
  if (tid < 64) { float s = r1ba[tid]; for (int n = 0; n < 16; ++n) s = fmaf(fmaxf(h[n], 0.f), r1wa[n * 64 + tid], s); tb[tid] = fmaxf(s, 0.f); }
  __syncthreads();
  if (tid < 16) { float s = r1bb[tid]; for (int o = 0; o < 64; ++o) s = fmaf(tb[o], r1wb[o * 16 + tid], s); h[tid] += s; }
  __syncthreads();
  if (tid < 32) { float s = b1[tid]; for (int n = 0; n < 16; ++n) s = fmaf(h[n], w1[n * 32 + tid], s); h1[tid] = fmaxf(s, 0.f); }
  __syncthreads();
  if (tid < 64) { float s = b2[tid]; for (int n = 0; n < 32; ++n) s = fmaf(h1[n], w2[n * 64 + tid], s); h2[tid] = fmaxf(s, 0.f); }
  __syncthreads();
  float o = b3[tid];
  for (int n = 0; n < 64; ++n) o = fmaf(h2[n], w3[n * 128 + tid], o);
  float sc = bsc[tid];
  for (int n = 0; n < 16; ++n) sc = fmaf(zq[n], wsc[n * 128 + tid], sc);
  out[(size_t)r * PLEN + tid] = o + sc;
}

// ---------------------------------------------------------------------------
// Launch.  Input order = JAX pytree flatten (dict keys sorted, lists in order):
//  0:x | enc: 1:b1 2:b2 3:b3 4:bsc 5-8:res0{ba,bb,wa,wb} 9-12:res1{..}
//           13:w1 14:w2 15:w3 16:wsc
//  dec: 17:b1 18:b2 19:b3 20:bsc 21-24:res0 25-28:res1 29:w1 30:w2 31:w3 32:wsc
//  33:codebook 34:cluster_size
// ---------------------------------------------------------------------------
extern "C" void kernel_launch(void* const* d_in, const int* in_sizes, int n_in,
                              void* d_out, int out_size, void* d_ws, size_t ws_size,
                              hipStream_t stream) {
  (void)in_sizes; (void)n_in; (void)out_size; (void)ws_size;
  const float* x   = (const float*)d_in[0];
  const float* eb1 = (const float*)d_in[1],  *eb2 = (const float*)d_in[2];
  const float* eb3 = (const float*)d_in[3],  *ebsc = (const float*)d_in[4];
  const float* e0ba = (const float*)d_in[5], *e0bb = (const float*)d_in[6];
  const float* e0wa = (const float*)d_in[7], *e0wb = (const float*)d_in[8];
  const float* e1ba = (const float*)d_in[9], *e1bb = (const float*)d_in[10];
  const float* e1wa = (const float*)d_in[11],*e1wb = (const float*)d_in[12];
  const float* ew1 = (const float*)d_in[13], *ew2 = (const float*)d_in[14];
  const float* ew3 = (const float*)d_in[15], *ewsc = (const float*)d_in[16];
  const float* db1 = (const float*)d_in[17], *db2 = (const float*)d_in[18];
  const float* db3 = (const float*)d_in[19], *dbsc = (const float*)d_in[20];
  const float* d0ba = (const float*)d_in[21],*d0bb = (const float*)d_in[22];
  const float* d0wa = (const float*)d_in[23],*d0wb = (const float*)d_in[24];
  const float* d1ba = (const float*)d_in[25],*d1bb = (const float*)d_in[26];
  const float* d1wa = (const float*)d_in[27],*d1wb = (const float*)d_in[28];
  const float* dw1 = (const float*)d_in[29], *dw2 = (const float*)d_in[30];
  const float* dw3 = (const float*)d_in[31], *dwsc = (const float*)d_in[32];
  const float* codebook = (const float*)d_in[33];
  const float* cluster_size = (const float*)d_in[34];

  // workspace carve-up
  float* z      = (float*)d_ws;                       // 917504
  float* meanv  = z + (size_t)NROWS * PLEN;           // 64
  float* rstdv  = meanv + LPP;                        // 64
  float* cbn2h  = rstdv + LPP;                        // 8192
  float* counts = cbn2h + KCODES;                     // 8192
  int*   idx_ws = (int*)(counts + KCODES);            // 7168
  _Float16* zh16 = (_Float16*)(idx_ws + NROWS);       // 7168*16 halves
  _Float16* cb16 = zh16 + (size_t)NROWS * HDIM;       // 8192*16 halves

  // output carve-up: [z_q_norm | idx | new_cluster_size]
  float* out_z   = (float*)d_out;
  float* out_idx = out_z + (size_t)NROWS * PLEN;
  float* out_cs  = out_idx + NROWS;

  fft_abs_kernel<<<NROWS, 128, 0, stream>>>(x, z);
  stats_kernel<<<LPP, 256, 0, stream>>>(z, meanv, rstdv);
  encoder_kernel<<<NROWS, 128, 0, stream>>>(z, meanv, rstdv,
      ew1, eb1, ew2, eb2, ew3, eb3,
      e0wa, e0ba, e0wb, e0bb, e1wa, e1ba, e1wb, e1bb,
      ewsc, ebsc, zh16);
  cbprep_kernel<<<KCODES / 256, 256, 0, stream>>>(codebook, cb16, cbn2h);
  cdist_argmin_kernel<<<NGROUP / 8, 256, 0, stream>>>(zh16, cb16, cbn2h, idx_ws, out_idx);
  hipMemsetAsync(counts, 0, KCODES * sizeof(float), stream);
  bincount_kernel<<<(NROWS + 255) / 256, 256, 0, stream>>>(idx_ws, counts);
  ema_kernel<<<KCODES / 256, 256, 0, stream>>>(cluster_size, counts, out_cs);
  decoder_kernel<<<NROWS, 128, 0, stream>>>(idx_ws, codebook,
      d0wa, d0ba, d0wb, d0bb, d1wa, d1ba, d1wb, d1bb,
      dw1, db1, dw2, db2, dw3, db3, dwsc, dbsc, out_z);
}
